// HeteroGraphSAGE_13434657702128
// MI455X (gfx1250) — compile-verified, matrix operand order checked by reading, not run on hardware
//
#include <hip/hip_runtime.h>
#include <hip/hip_bf16.h>

// ---------------------------------------------------------------------------
// HeteroGraphSAGE on MI455X (gfx1250, wave32).
// bf16 activations, fp32 atomic scatter-mean (L2-resident), WMMA bf16 GEMMs
// with 64-row M tiles (4x B-fragment register reuse) and fused mean/convert.
// ---------------------------------------------------------------------------

typedef __bf16 bf16_t;
typedef __attribute__((ext_vector_type(16))) __bf16 v16bf;
typedef __attribute__((ext_vector_type(8)))  __bf16 v8bf;
typedef __attribute__((ext_vector_type(4)))  __bf16 v4bf;
typedef __attribute__((ext_vector_type(8)))  float  v8f;

#define HDIM 128
#define NCN  50000
#define NMN  30000
#define NDN  10000
#define ECM  800000
#define EMD  400000
#define ECD  800000

// ---------------- fp32 -> bf16 (vectorized x4) -----------------------------
__global__ void k_f32_to_bf16(const float* __restrict__ x, bf16_t* __restrict__ y, int n4) {
  int i = blockIdx.x * blockDim.x + threadIdx.x;
  if (i < n4) {
    float4 v = ((const float4*)x)[i];
    v4bf o;
    o[0] = (bf16_t)v.x; o[1] = (bf16_t)v.y; o[2] = (bf16_t)v.z; o[3] = (bf16_t)v.w;
    ((v4bf*)y)[i] = o;
  }
}

// ---------------- weight prep: transpose (K-major -> N-major) + bf16 -------
// Wt[(l*6+r)*2 + which][n][k] = W[l][r][k][n]
__global__ void k_prep_weights(const float* __restrict__ Wl, const float* __restrict__ Wr,
                               bf16_t* __restrict__ Wt) {
  int mat = blockIdx.x;  // 0..23 = l*6+r
  const float* srcs[2] = {Wl + (size_t)mat * HDIM * HDIM, Wr + (size_t)mat * HDIM * HDIM};
  for (int which = 0; which < 2; ++which) {
    bf16_t* out = Wt + (size_t)(mat * 2 + which) * HDIM * HDIM;
    const float* in = srcs[which];
    for (int id = threadIdx.x; id < HDIM * HDIM; id += blockDim.x) {
      int n = id >> 7, k = id & 127;
      out[id] = (bf16_t)in[k * HDIM + n];
    }
  }
}

// ---------------- scatter-add: one wave per edge ---------------------------
__global__ void k_scatter(const bf16_t* __restrict__ hsrc,
                          const int* __restrict__ src, const int* __restrict__ dst,
                          int nE, float* __restrict__ agg, float* __restrict__ cnt) {
  int lane = threadIdx.x & 31;
  int wave = (blockIdx.x * blockDim.x + threadIdx.x) >> 5;
  int nWaves = (gridDim.x * blockDim.x) >> 5;
  for (int e0 = wave; e0 < nE; e0 += nWaves) {
    int e = __builtin_amdgcn_readfirstlane(e0);     // wave-uniform -> s_load indices
    int s = src[e];
    int d = dst[e];
    const bf16_t* row = hsrc + (size_t)s * HDIM;
    float* out = agg + (size_t)d * HDIM;
    v4bf v = *(const v4bf*)(row + lane * 4);        // coalesced 8B/lane gather
#pragma unroll
    for (int i = 0; i < 4; ++i)
      __hip_atomic_fetch_add(out + lane * 4 + i, (float)v[i],
                             __ATOMIC_RELAXED, __HIP_MEMORY_SCOPE_AGENT);
    if (lane == 0)
      __hip_atomic_fetch_add(cnt + d, 1.0f, __ATOMIC_RELAXED, __HIP_MEMORY_SCOPE_AGENT);
  }
}

// ---------------- fused WMMA GEMM ------------------------------------------
// out = act( 0.5*( meanA@WlA + meanB@WlB + h@WrA + h@WrB + bA + bB ) )
// meanX = aggX / max(cntX,1) is fused into LDS staging (fp32 -> bf16).
// Block = 256 threads = 8 waves; wave owns one 16-wide N tile; M tile = 64
// rows (4 sub-tiles): each B fragment is register-reused across 4 WMMAs.
__launch_bounds__(256)
__global__ void k_gemm_fused(const float* __restrict__ aggA, const float* __restrict__ cntA,
                             const bf16_t* __restrict__ WlA,
                             const float* __restrict__ aggB, const float* __restrict__ cntB,
                             const bf16_t* __restrict__ WlB,
                             const bf16_t* __restrict__ hroot,
                             const bf16_t* __restrict__ WrA, const bf16_t* __restrict__ WrB,
                             const float* __restrict__ biasA, const float* __restrict__ biasB,
                             int nRows, int doRelu,
                             bf16_t* __restrict__ outBf, float* __restrict__ outF) {
  __shared__ __align__(16) bf16_t lA[3][64][HDIM];   // 48 KB
  int tid  = threadIdx.x;
  int row0 = blockIdx.x * 64;

  // ---- cooperative staging: 4 threads per row, quarter-row each ----
  {
    int r = tid >> 2;          // 0..63
    int q = tid & 3;           // row quarter (32 cols)
    int grow = row0 + r;
    bool inb = grow < nRows;
    const float* aggs[2] = {aggA, aggB};
    const float* cnts[2] = {cntA, cntB};
#pragma unroll
    for (int s = 0; s < 2; ++s) {
      float inv = 0.0f;
      if (inb) inv = 1.0f / fmaxf(cnts[s][grow], 1.0f);
      const float* rowp = aggs[s] + (size_t)grow * HDIM;
#pragma unroll
      for (int j = 0; j < 8; ++j) {
        int col = q * 32 + j * 4;
        float4 v = make_float4(0.f, 0.f, 0.f, 0.f);
        if (inb) v = *(const float4*)(rowp + col);
        v4bf o;
        o[0] = (bf16_t)(v.x * inv); o[1] = (bf16_t)(v.y * inv);
        o[2] = (bf16_t)(v.z * inv); o[3] = (bf16_t)(v.w * inv);
        *(v4bf*)(&lA[s][r][col]) = o;
      }
    }
    const bf16_t* rowp = hroot + (size_t)grow * HDIM;
#pragma unroll
    for (int j = 0; j < 4; ++j) {
      int col = q * 32 + j * 8;
      v8bf v = {};
      if (inb) v = *(const v8bf*)(rowp + col);
      *(v8bf*)(&lA[2][r][col]) = v;
    }
  }
  __syncthreads();

  int lane = tid & 31;
  int wv   = tid >> 5;           // N tile 0..7
  int nl   = lane & 15;          // M index of A frag / N index of B frag
  int kg   = lane >> 4;          // half-wave K group

  const bf16_t* Bmat[4] = {WlA, WlB, WrA, WrB};
  const int Asel[4] = {0, 1, 2, 2};

  v8f acc[4] = {};
#pragma unroll
  for (int p = 0; p < 4; ++p) {
    const bf16_t* Brow = Bmat[p] + (size_t)(wv * 16 + nl) * HDIM;  // N-major weights
#pragma unroll
    for (int kk = 0; kk < 4; ++kk) {
      // ISA 16-bit 16x32 layout: lane group kg covers K = kb..kb+7, kb+16..kb+23
      int kb = kk * 32 + kg * 8;
      v8bf b0 = *(const v8bf*)(Brow + kb);
      v8bf b1 = *(const v8bf*)(Brow + kb + 16);
      v16bf b;
#pragma unroll
      for (int i = 0; i < 8; ++i) { b[i] = b0[i]; b[i + 8] = b1[i]; }
#pragma unroll
      for (int s4 = 0; s4 < 4; ++s4) {             // 4 M sub-tiles reuse b
        const bf16_t* Arow = &lA[Asel[p]][s4 * 16 + nl][0];
        v8bf a0 = *(const v8bf*)(Arow + kb);
        v8bf a1 = *(const v8bf*)(Arow + kb + 16);
        v16bf a;
#pragma unroll
        for (int i = 0; i < 8; ++i) { a[i] = a0[i]; a[i + 8] = a1[i]; }
        acc[s4] = __builtin_amdgcn_wmma_f32_16x16x32_bf16(false, a, false, b,
                                                          (short)0, acc[s4], false, false);
      }
    }
  }

  int n = wv * 16 + nl;
  float bias = biasA[n] + biasB[n];
#pragma unroll
  for (int s4 = 0; s4 < 4; ++s4) {
#pragma unroll
    for (int i = 0; i < 8; ++i) {
      int row = row0 + s4 * 16 + kg * 8 + i;       // C layout: VGPR i -> M=i+8*kg
      if (row < nRows) {
        float v = 0.5f * (acc[s4][i] + bias);
        if (doRelu) v = fmaxf(v, 0.0f);
        if (outF) outF[(size_t)row * HDIM + n] = v;
        else      outBf[(size_t)row * HDIM + n] = (bf16_t)v;
      }
    }
  }
}

// ---------------- row L2 normalize (wave per row) --------------------------
__global__ void k_l2norm(float* __restrict__ x, int nRows) {
  int lane = threadIdx.x & 31;
  int wave = (blockIdx.x * blockDim.x + threadIdx.x) >> 5;
  if (wave >= nRows) return;
  float4 v = *(const float4*)(x + (size_t)wave * HDIM + lane * 4);
  float ss = v.x * v.x + v.y * v.y + v.z * v.z + v.w * v.w;
#pragma unroll
  for (int off = 16; off > 0; off >>= 1) ss += __shfl_xor(ss, off, 32);
  float inv = 1.0f / fmaxf(sqrtf(ss), 1e-12f);
  v.x *= inv; v.y *= inv; v.z *= inv; v.w *= inv;
  *(float4*)(x + (size_t)wave * HDIM + lane * 4) = v;
}

// ---------------------------------------------------------------------------
extern "C" void kernel_launch(void* const* d_in, const int* in_sizes, int n_in,
                              void* d_out, int out_size, void* d_ws, size_t ws_size,
                              hipStream_t stream) {
  (void)in_sizes; (void)n_in; (void)out_size; (void)ws_size;

  const float* x_c  = (const float*)d_in[0];
  const float* x_m  = (const float*)d_in[1];
  const float* x_d  = (const float*)d_in[2];
  const float* Wl   = (const float*)d_in[15];
  const float* Wr   = (const float*)d_in[16];
  const float* bias = (const float*)d_in[17];

  // ---- workspace carve-out (256B aligned) ----
  char* p = (char*)d_ws;
  auto take = [&](size_t bytes) -> char* {
    char* r = p; p += (bytes + 255) & ~(size_t)255; return r;
  };
  const size_t NTOT = (size_t)NCN + NMN + NDN;
  bf16_t* hb0   = (bf16_t*)take(NTOT * HDIM * sizeof(bf16_t));
  bf16_t* hb1   = (bf16_t*)take(NTOT * HDIM * sizeof(bf16_t));
  bf16_t* hb[2] = {hb0, hb1};
  float*  aggA  = (float*) take((size_t)NCN * HDIM * sizeof(float));
  float*  aggB  = (float*) take((size_t)NCN * HDIM * sizeof(float));
  float*  cntA  = (float*) take((size_t)NCN * sizeof(float));
  float*  cntB  = (float*) take((size_t)NCN * sizeof(float));
  bf16_t* Wt    = (bf16_t*)take((size_t)48 * HDIM * HDIM * sizeof(bf16_t));

  // ---- one-time prep ----
  k_prep_weights<<<24, 256, 0, stream>>>(Wl, Wr, Wt);
  const int offC = 0, offM = NCN * HDIM, offD = (NCN + NMN) * HDIM;
  k_f32_to_bf16<<<(NCN * HDIM / 4 + 255) / 256, 256, 0, stream>>>(x_c, hb[0] + offC, NCN * HDIM / 4);
  k_f32_to_bf16<<<(NMN * HDIM / 4 + 255) / 256, 256, 0, stream>>>(x_m, hb[0] + offM, NMN * HDIM / 4);
  k_f32_to_bf16<<<(NDN * HDIM / 4 + 255) / 256, 256, 0, stream>>>(x_d, hb[0] + offD, NDN * HDIM / 4);

  // ---- per-dst-type relation descriptors ----
  struct RelD { const int* src; const int* dst; int nE; int srcOff; };
  struct DstD { int n; int hOff; RelD a; RelD b; int ra; int rb; };
  DstD dts[3] = {
    // c <- (m->c, r=3), (d->c, r=5)
    {NCN, offC, {(const int*)d_in[9],  (const int*)d_in[10], ECM, offM},
                {(const int*)d_in[13], (const int*)d_in[14], ECD, offD}, 3, 5},
    // m <- (c->m, r=0), (d->m, r=4)
    {NMN, offM, {(const int*)d_in[3],  (const int*)d_in[4],  ECM, offC},
                {(const int*)d_in[11], (const int*)d_in[12], EMD, offD}, 0, 4},
    // d <- (m->d, r=1), (c->d, r=2)
    {NDN, offD, {(const int*)d_in[5],  (const int*)d_in[6],  EMD, offM},
                {(const int*)d_in[7],  (const int*)d_in[8],  ECD, offC}, 1, 2},
  };

  int cur = 0;
  for (int l = 0; l < 4; ++l) {
    int nxt = cur ^ 1;
    for (int t = 0; t < 3; ++t) {
      DstD& D = dts[t];
      hipMemsetAsync(aggA, 0, (size_t)D.n * HDIM * sizeof(float), stream);
      hipMemsetAsync(cntA, 0, (size_t)D.n * sizeof(float), stream);
      hipMemsetAsync(aggB, 0, (size_t)D.n * HDIM * sizeof(float), stream);
      hipMemsetAsync(cntB, 0, (size_t)D.n * sizeof(float), stream);
      k_scatter<<<4096, 256, 0, stream>>>(hb[cur] + D.a.srcOff, D.a.src, D.a.dst, D.a.nE, aggA, cntA);
      k_scatter<<<4096, 256, 0, stream>>>(hb[cur] + D.b.srcOff, D.b.src, D.b.dst, D.b.nE, aggB, cntB);

      const bf16_t* WlAp = Wt + (size_t)((l * 6 + D.ra) * 2 + 0) * HDIM * HDIM;
      const bf16_t* WrAp = Wt + (size_t)((l * 6 + D.ra) * 2 + 1) * HDIM * HDIM;
      const bf16_t* WlBp = Wt + (size_t)((l * 6 + D.rb) * 2 + 0) * HDIM * HDIM;
      const bf16_t* WrBp = Wt + (size_t)((l * 6 + D.rb) * 2 + 1) * HDIM * HDIM;
      const float*  bA   = bias + (size_t)(l * 6 + D.ra) * HDIM;
      const float*  bB   = bias + (size_t)(l * 6 + D.rb) * HDIM;

      bf16_t* outBf = (l < 3) ? (hb[nxt] + D.hOff) : nullptr;
      float*  outF  = (l == 3) ? ((float*)d_out + D.hOff) : nullptr;
      k_gemm_fused<<<(D.n + 63) / 64, 256, 0, stream>>>(
          aggA, cntA, WlAp, aggB, cntB, WlBp, hb[cur] + D.hOff, WrAp, WrBp,
          bA, bB, D.n, (l < 3) ? 1 : 0, outBf, outF);
    }
    cur = nxt;
  }

  int totRows = NCN + NMN + NDN;
  k_l2norm<<<((size_t)totRows * 32 + 255) / 256, 256, 0, stream>>>((float*)d_out, totRows);
}
